// QuantizerEMA_18485539242753
// MI455X (gfx1250) — matrix operand order repaired and checked by reading, not compiled, not run on hardware
//
#include <hip/hip_runtime.h>
#include <math.h>

#define DD 128
#define KK 4096
#define NROWS 65536
#define NT (KK / 16)          // 256 k-tiles
#define DECAYF 0.99f
#define OMDF 0.01f
#define EPSF 1e-5f

typedef __bf16 bf16;
typedef __attribute__((ext_vector_type(16))) __bf16 v16bf;
typedef __attribute__((ext_vector_type(8)))  float v8f;

union AFrag { v16bf v; bf16 h[16]; };
union BFrag { v16bf v; uint4 q[2]; };
union CFrag { v8f  v; float f[8];  };

#if __has_builtin(__builtin_amdgcn_global_load_async_to_lds_b128) && \
    __has_builtin(__builtin_amdgcn_s_wait_asynccnt)
#define HAS_ASYNC_LDS 1
typedef int v4i __attribute__((vector_size(16)));
typedef __attribute__((address_space(1))) v4i g_v4i;   // global (prints as __device__)
typedef __attribute__((address_space(3))) v4i l_v4i;   // LDS    (prints as __shared__)
#else
#define HAS_ASYNC_LDS 0
#endif

// ---- prep: repack E (f32 [D,K]) into bf16 B-fragment-major layout -------------
// For ktile kt, chunk c, lane l (half=l>>4, n=l&15), slot j (0..15):
//   value = E[c*32 + 16*half + j, kt*16 + n]
// stored at EbfFrag[ ((kt*4 + c)*32 + l)*16 + j ]  -> 4KB contiguous per ktile.
__global__ void vq_pack_b(const float* __restrict__ E, bf16* __restrict__ EbfFrag) {
    int i = blockIdx.x * blockDim.x + threadIdx.x;   // DD*KK threads
    int j  = i & 15;
    int l  = (i >> 4) & 31;
    int c  = (i >> 9) & 3;
    int kt = i >> 11;
    int half = l >> 4, n = l & 15;
    int d   = c * 32 + 16 * half + j;
    int col = kt * 16 + n;
    EbfFrag[i] = (bf16)E[(size_t)d * KK + col];
}

// ---- prep: colnorm[k] = sum_d E[d,k]^2 ----------------------------------------
__global__ void vq_colnorm(const float* __restrict__ E, float* __restrict__ colnorm) {
    int k = blockIdx.x * blockDim.x + threadIdx.x;   // KK threads
    float s = 0.f;
    #pragma unroll 4
    for (int d = 0; d < DD; ++d) {
        float e = E[d * KK + k];
        s += e * e;
    }
    colnorm[k] = s;
}

// ---- main: WMMA distance + argmin ---------------------------------------------
// block = 256 threads = 8 waves; each wave owns a 16-row tile; grid = NROWS/128
__global__ __launch_bounds__(256) void vq_dist_argmin(
    const float* __restrict__ x, const bf16* __restrict__ EbfFrag,
    const float* __restrict__ colnorm, int* __restrict__ idxOut)
{
    __shared__ bf16 Et[2][2048];   // double-buffered 4KB fragment-major E tiles

    const int tid  = threadIdx.x;
    const int wave = tid >> 5;
    const int lane = tid & 31;
    const int half = lane >> 4;
    const int hlan = lane & 15;
    const int rowBase = blockIdx.x * 128 + wave * 16;
    const int myRow   = rowBase + hlan;

    // Build A fragments (16x32 bf16, contiguous k = 16*half + s) for the 4 D-chunks.
    AFrag a[4];
    #pragma unroll
    for (int c = 0; c < 4; ++c) {
        #pragma unroll
        for (int i = 0; i < 8; ++i) {
            int ka = 2 * i + 16 * half;
            const float2 p = *(const float2*)(x + (size_t)myRow * DD + c * 32 + ka);
            a[c].h[2 * i]     = (bf16)p.x;
            a[c].h[2 * i + 1] = (bf16)p.y;
        }
    }

    float minv[8];
    int   mini[8];
    #pragma unroll
    for (int v = 0; v < 8; ++v) { minv[v] = 3.4e38f; mini[v] = 0; }

    // staging: 256 threads x 16B = 4KB per tile, fully coalesced
    const bf16* srcBase = EbfFrag + (size_t)tid * 8;   // + kt*2048
    bf16* dst0 = &Et[0][tid * 8];
    bf16* dst1 = &Et[1][tid * 8];

#if HAS_ASYNC_LDS
    __builtin_amdgcn_global_load_async_to_lds_b128(
        (g_v4i*)(void*)srcBase, (l_v4i*)(void*)dst0, 0, 0);
    __builtin_amdgcn_s_wait_asynccnt(0);
#else
    *(uint4*)dst0 = *(const uint4*)srcBase;
#endif
    __syncthreads();

    for (int kt = 0; kt < NT; ++kt) {
        const int cur = kt & 1;
        const bf16* Ecur = Et[cur];
        bf16* Enext = cur ? dst0 : dst1;

#if HAS_ASYNC_LDS
        if (kt + 1 < NT)
            __builtin_amdgcn_global_load_async_to_lds_b128(
                (g_v4i*)(void*)(srcBase + (size_t)(kt + 1) * 2048),
                (l_v4i*)(void*)Enext, 0, 0);
#else
        uint4 pf;
        if (kt + 1 < NT)
            pf = *(const uint4*)(srcBase + (size_t)(kt + 1) * 2048);
#endif

        CFrag acc;
        #pragma unroll
        for (int v = 0; v < 8; ++v) acc.f[v] = 0.f;

        #pragma unroll
        for (int c = 0; c < 4; ++c) {
            BFrag b;   // this lane's 16 bf16 B operand: 32 contiguous bytes in LDS
            const uint4* bp = (const uint4*)(Ecur + (size_t)(c * 32 + lane) * 16);
            b.q[0] = bp[0];
            b.q[1] = bp[1];
            acc.v = __builtin_amdgcn_wmma_f32_16x16x32_bf16(
                        false, a[c].v, false, b.v, (short)0, acc.v, false, false);
        }

        const float cn   = colnorm[kt * 16 + hlan];
        const int   kidx = kt * 16 + hlan;
        #pragma unroll
        for (int v = 0; v < 8; ++v) {
            float dval = cn - 2.0f * acc.f[v];   // ||E||^2 - 2 x.E  (||x||^2 const)
            if (dval < minv[v]) { minv[v] = dval; mini[v] = kidx; }
        }

#if HAS_ASYNC_LDS
        if (kt + 1 < NT)
            __builtin_amdgcn_s_wait_asynccnt(0);
#else
        if (kt + 1 < NT)
            *(uint4*)Enext = pf;
#endif
        __syncthreads();
    }

    // reduce (min,idx) across the 16 lanes of each half (wave32 xor shuffles)
    #pragma unroll
    for (int off = 1; off < 16; off <<= 1) {
        #pragma unroll
        for (int v = 0; v < 8; ++v) {
            float ov = __shfl_xor(minv[v], off, 32);
            int   oi = __shfl_xor(mini[v], off, 32);
            if (ov < minv[v] || (ov == minv[v] && oi < mini[v])) {
                minv[v] = ov; mini[v] = oi;
            }
        }
    }
    if (hlan == 0) {
        #pragma unroll
        for (int v = 0; v < 8; ++v)
            idxOut[rowBase + v + 8 * half] = mini[v];   // row = vgpr + 8*half
    }
}

// ---- scatter: counts + x^T @ onehot via f32 atomics ---------------------------
__global__ void vq_scatter(const float* __restrict__ x, const int* __restrict__ idx,
                           float* __restrict__ counts, float* __restrict__ dwacc) {
    int t = blockIdx.x * blockDim.x + threadIdx.x;   // NROWS*DD threads
    int n = t >> 7, d = t & 127;
    int k = idx[n];
    atomicAdd(&dwacc[(size_t)d * KK + k], x[t]);
    if (d == 0) atomicAdd(&counts[k], 1.0f);
}

// ---- cluster EMA + reductions (n_sum, perplexity sum) -------------------------
__global__ void vq_cluster(const float* __restrict__ ema_cs, const float* __restrict__ counts,
                           float* __restrict__ out_cluster, float* __restrict__ scal) {
    __shared__ float red[256];
    int k = blockIdx.x * 256 + threadIdx.x;          // KK threads
    float nc = ema_cs[k] * DECAYF + counts[k] * OMDF;
    out_cluster[k] = nc;
    float p  = counts[k] * (1.0f / (float)NROWS);
    float pl = p * logf(p + 1e-10f);

    red[threadIdx.x] = nc;
    __syncthreads();
    for (int s = 128; s > 0; s >>= 1) {
        if (threadIdx.x < s) red[threadIdx.x] += red[threadIdx.x + s];
        __syncthreads();
    }
    if (threadIdx.x == 0) atomicAdd(&scal[0], red[0]);
    __syncthreads();

    red[threadIdx.x] = pl;
    __syncthreads();
    for (int s = 128; s > 0; s >>= 1) {
        if (threadIdx.x < s) red[threadIdx.x] += red[threadIdx.x + s];
        __syncthreads();
    }
    if (threadIdx.x == 0) atomicAdd(&scal[1], red[0]);
}

// ---- dw EMA + normalized embedding --------------------------------------------
__global__ void vq_emb(const float* __restrict__ ema_dw, const float* __restrict__ dwacc,
                       const float* __restrict__ out_cluster, const float* __restrict__ scal,
                       float* __restrict__ out_new_dw, float* __restrict__ out_new_emb) {
    int i = blockIdx.x * blockDim.x + threadIdx.x;   // DD*KK threads
    int k = i & (KK - 1);
    float ndw = ema_dw[i] * DECAYF + dwacc[i] * OMDF;
    out_new_dw[i] = ndw;
    float nsum  = scal[0];
    float denom = nsum * (out_cluster[k] + EPSF) / (nsum + (float)KK * EPSF);
    out_new_emb[i] = ndw / denom;
}

// ---- gather q + commitment loss reduction -------------------------------------
__global__ void vq_gather(const float* __restrict__ x, const int* __restrict__ idx,
                          const float* __restrict__ new_emb,
                          float* __restrict__ out_q, float* __restrict__ scal) {
    __shared__ float red[256];
    int t = blockIdx.x * 256 + threadIdx.x;          // NROWS*DD threads
    int n = t >> 7, d = t & 127;
    float qv = new_emb[(size_t)d * KK + idx[n]];
    out_q[t] = qv;
    float df = qv - x[t];
    red[threadIdx.x] = df * df;
    __syncthreads();
    for (int s = 128; s > 0; s >>= 1) {
        if (threadIdx.x < s) red[threadIdx.x] += red[threadIdx.x + s];
        __syncthreads();
    }
    if (threadIdx.x == 0) atomicAdd(&scal[2], red[0]);
}

// ---- scalars ------------------------------------------------------------------
__global__ void vq_final(const float* __restrict__ scal,
                         float* __restrict__ out_loss, float* __restrict__ out_perp) {
    if (threadIdx.x == 0 && blockIdx.x == 0) {
        out_loss[0] = scal[2] * (1.0f / (float)((size_t)NROWS * DD));
        out_perp[0] = expf(-scal[1]);
    }
}

extern "C" void kernel_launch(void* const* d_in, const int* in_sizes, int n_in,
                              void* d_out, int out_size, void* d_ws, size_t ws_size,
                              hipStream_t stream) {
    const float* x       = (const float*)d_in[0];   // [NROWS, DD]
    const float* E       = (const float*)d_in[1];   // [DD, KK]
    const float* ema_cs  = (const float*)d_in[2];   // [KK]
    const float* ema_dw  = (const float*)d_in[3];   // [DD, KK]

    float* out      = (float*)d_out;
    float* out_q    = out;                  // 8388608
    float* out_loss = out + 8388608;        // 1
    float* out_perp = out + 8388609;        // 1
    float* out_emb  = out + 8388610;        // 524288
    float* out_clu  = out + 8912898;        // 4096
    float* out_dw   = out + 8916994;        // 524288

    char* ws = (char*)d_ws;
    bf16*  EbfFrag = (bf16*)ws;   ws += (size_t)DD * KK * sizeof(bf16);   // 1 MB
    float* colnorm = (float*)ws;  ws += (size_t)KK * sizeof(float);
    float* counts  = (float*)ws;  ws += (size_t)KK * sizeof(float);
    float* dwacc   = (float*)ws;  ws += (size_t)DD * KK * sizeof(float);  // 2 MB
    float* scal    = (float*)ws;  ws += 4 * sizeof(float);                // n, perpsum, losssum
    int*   idx     = (int*)ws;                                            // [NROWS]

    (void)hipMemsetAsync(counts, 0, (size_t)KK * sizeof(float), stream);
    (void)hipMemsetAsync(dwacc,  0, (size_t)DD * KK * sizeof(float), stream);
    (void)hipMemsetAsync(scal,   0, 4 * sizeof(float), stream);

    vq_pack_b    <<<(DD * KK) / 256, 256, 0, stream>>>(E, EbfFrag);
    vq_colnorm   <<<KK / 256,        256, 0, stream>>>(E, colnorm);
    vq_dist_argmin<<<NROWS / 128,    256, 0, stream>>>(x, EbfFrag, colnorm, idx);
    vq_scatter   <<<((size_t)NROWS * DD) / 256, 256, 0, stream>>>(x, idx, counts, dwacc);
    vq_cluster   <<<KK / 256,        256, 0, stream>>>(ema_cs, counts, out_clu, scal);
    vq_emb       <<<(DD * KK) / 256, 256, 0, stream>>>(ema_dw, dwacc, out_clu, scal,
                                                       out_dw, out_emb);
    vq_gather    <<<((size_t)NROWS * DD) / 256, 256, 0, stream>>>(x, idx, out_emb,
                                                                  out_q, scal);
    vq_final     <<<1, 32, 0, stream>>>(scal, out_loss, out_perp);
}
